// CrossAttentionScaler_62440234549492
// MI455X (gfx1250) — matrix operand order repaired and checked by reading, not compiled, not run on hardware
//
#include <hip/hip_runtime.h>

// ---------------------------------------------------------------------------
// CDNA5 (gfx1250) wave32 WMMA implementation of CrossAttention + Transitions.
// All matmuls: v_wmma_f32_16x16x32_f16 (f16 A/B, f32 accumulate).
// Attention: flash-style streaming softmax; logits never hit memory.
// All fragment loaders are branch-free (padding handled by explicit repack).
// ---------------------------------------------------------------------------

typedef __attribute__((ext_vector_type(16))) _Float16 v16h;
typedef __attribute__((ext_vector_type(2)))  _Float16 h2_t;
typedef __attribute__((ext_vector_type(8)))  float    v8f;

#define WMMA_F16(a, b, c) \
  __builtin_amdgcn_wmma_f32_16x16x32_f16(false, (a), false, (b), (short)0, (c), false, false)

static constexpr int T_TOK = 512;
static constexpr int R_RES = 4096;
static constexpr int C_DIM = 384;
static constexpr int H_HEAD = 8;
static constexpr int KD = 48;
static constexpr int KD_PAD = 64;                 // per-head padded kd
static constexpr int C_PAD = H_HEAD * KD_PAD;     // 512
static constexpr int F_DIM = 1536;
static constexpr float BIGF = 1e9f;

__device__ __forceinline__ int lane_id() { return (int)(threadIdx.x & 31u); }

// --- A fragment: 16x32 f16 tile of row-major matrix [*, lda], branch-free ----
// Layout used here (16-bit A 16x32): lane L: m = L&15, half = L>>4;
//   VGPR j holds K = half*16 + 2j, +1 (contiguous order, matches B)
__device__ __forceinline__ v16h load_a_rm(const _Float16* A, int lda, int row0,
                                          int kcol0) {
  const int l = lane_id();
  const int m = l & 15, half = l >> 4;
  const _Float16* rp = A + (size_t)(row0 + m) * lda + kcol0;
  v16h a;
#pragma unroll
  for (int j = 0; j < 8; ++j) {
    const int kb = half * 16 + 2 * j;
    const h2_t p = *(const h2_t*)(rp + kb);       // 32-bit paired load
    a[2 * j] = p[0]; a[2 * j + 1] = p[1];
  }
  return a;
}

// --- B fragment where B = X^T, X row-major [N, ldx]: B(k,n) = X[n,k] ---------
// Layout used here (16-bit B 32x16): lane L: n = L&15, half = L>>4;
//   VGPR j holds K = half*16 + 2j, +1 for column n.  K-pairs contiguous in X.
__device__ __forceinline__ v16h load_b_tr(const _Float16* X, int ldx, int krow0,
                                          int col0) {
  const int l = lane_id();
  const int n = l & 15, half = l >> 4;
  const _Float16* rp = X + (size_t)(col0 + n) * ldx + krow0 + half * 16;
  v16h b;
#pragma unroll
  for (int j = 0; j < 8; ++j) {
    const h2_t p = *(const h2_t*)(rp + 2 * j);    // 32-bit paired load
    b[2 * j] = p[0]; b[2 * j + 1] = p[1];
  }
  return b;
}

// ---------------------------------------------------------------------------
// GEMM: C[M,N] = act(alpha * A[M,K] @ B[K,N] + bias) + residual.
// B is passed TRANSPOSED (BT[N,K], row-major) so fragment pair-loads are
// contiguous. One wave computes a 16x64 output strip (A fragment reused 4x).
// ---------------------------------------------------------------------------
__global__ __launch_bounds__(256)
void gemm_f16_kernel(const _Float16* __restrict__ A, int lda,
                     const _Float16* __restrict__ BT,   // [N, K]
                     const float* __restrict__ bias,
                     const float* __restrict__ residual,
                     float alpha, int relu,
                     float* __restrict__ outF, _Float16* __restrict__ outH,
                     int M, int N, int K) {
  const int tilesN = N >> 6;                     // 64-wide strips
  const int total  = (M >> 4) * tilesN;
  const int wid = (int)blockIdx.x * (int)(blockDim.x >> 5) + (int)(threadIdx.x >> 5);
  if (wid >= total) return;                      // wave-uniform branch
  const int tm = (wid / tilesN) << 4;
  const int tn = (wid % tilesN) << 6;

  v8f acc0 = {}, acc1 = {}, acc2 = {}, acc3 = {};
  for (int k0 = 0; k0 < K; k0 += 32) {
    const v16h a = load_a_rm(A, lda, tm, k0);
    acc0 = WMMA_F16(a, load_b_tr(BT, K, k0, tn),      acc0);
    acc1 = WMMA_F16(a, load_b_tr(BT, K, k0, tn + 16), acc1);
    acc2 = WMMA_F16(a, load_b_tr(BT, K, k0, tn + 32), acc2);
    acc3 = WMMA_F16(a, load_b_tr(BT, K, k0, tn + 48), acc3);
  }

  const int l = lane_id();
  const int n = l & 15, half = l >> 4;
  v8f accs[4] = {acc0, acc1, acc2, acc3};
#pragma unroll
  for (int s = 0; s < 4; ++s) {
    const int col = tn + s * 16 + n;
    const float bv = bias ? bias[col] : 0.f;
#pragma unroll
    for (int i = 0; i < 8; ++i) {
      const int row = tm + half * 8 + i;
      float v = accs[s][i] * alpha + bv;
      if (relu) v = fmaxf(v, 0.f);
      if (residual) v += residual[(size_t)row * N + col];
      const size_t o = (size_t)row * N + col;
      if (outF) outF[o] = v;
      if (outH) outH[o] = (_Float16)v;
    }
  }
}

// ---------------------------------------------------------------------------
// LayerNorm (fp32 in, f16 out). One block per row of length C.
// ---------------------------------------------------------------------------
__global__ __launch_bounds__(128)
void layernorm_f16_kernel(const float* __restrict__ x,
                          const float* __restrict__ scale,
                          const float* __restrict__ offset,
                          _Float16* __restrict__ out, int C) {
  const int row = blockIdx.x;
  const float* xr = x + (size_t)row * C;
  float s = 0.f, s2 = 0.f;
  for (int c = threadIdx.x; c < C; c += blockDim.x) {
    const float v = xr[c]; s += v; s2 += v * v;
  }
#pragma unroll
  for (int off = 16; off; off >>= 1) {
    s  += __shfl_xor(s,  off, 32);
    s2 += __shfl_xor(s2, off, 32);
  }
  __shared__ float sh[8], sh2[8];
  const int w = threadIdx.x >> 5;
  if ((threadIdx.x & 31) == 0) { sh[w] = s; sh2[w] = s2; }
  __syncthreads();
  s = 0.f; s2 = 0.f;
  const int nw = blockDim.x >> 5;
  for (int i = 0; i < nw; ++i) { s += sh[i]; s2 += sh2[i]; }
  const float mu  = s / C;
  const float var = s2 / C - mu * mu;
  const float rs  = rsqrtf(var + 1e-5f);
  for (int c = threadIdx.x; c < C; c += blockDim.x) {
    out[(size_t)row * C + c] = (_Float16)((xr[c] - mu) * rs * scale[c] + offset[c]);
  }
}

// ---------------------------------------------------------------------------
// Flash attention: one wave per (head, 16-query tile). Streams R in chunks of
// 32 keys; online softmax; logits never hit memory. Mask read exactly once.
//   qp  : f16 [T, 512]  head-padded (kd 48 -> 64, zero pad), pre-scaled kd^-0.5
//   kp  : f16 [R, 512]  head-padded
//   vT  : f16 [384, R]  (head-major rows h*48+c)
//   mask: f32 [H, T, R]
//   outw: f32 [T, 384]
// ---------------------------------------------------------------------------
__global__ __launch_bounds__(32)
void flash_attn_kernel(const _Float16* __restrict__ qp,
                       const _Float16* __restrict__ kp,
                       const _Float16* __restrict__ vT,
                       const float* __restrict__ mask,
                       float* __restrict__ outw) {
  const int h  = blockIdx.y;
  const int t0 = blockIdx.x * 16;
  const int l  = lane_id();
  const int n = l & 15, half = l >> 4;

  // Q fragments for this 16-row tile (padded K = 64 -> 2 fragments)
  const v16h aq0 = load_a_rm(qp + h * KD_PAD, C_PAD, t0, 0);
  const v16h aq1 = load_a_rm(qp + h * KD_PAD, C_PAD, t0, 32);

  float mrow[8], lrow[8];
#pragma unroll
  for (int i = 0; i < 8; ++i) { mrow[i] = -1e30f; lrow[i] = 0.f; }
  v8f o0 = {}, o1 = {}, o2 = {};

  __shared__ _Float16 ptile[16 * 32];            // P tile staging (D->A layout)

  const float* mrowp = mask + ((size_t)h * T_TOK + t0) * R_RES;
  const _Float16* kh  = kp + h * KD_PAD;
  const _Float16* vTh = vT + (size_t)h * KD * R_RES;

  for (int r0 = 0; r0 < R_RES; r0 += 32) {
    if (r0 + 32 < R_RES)                          // gfx1250 global_prefetch_b8
      __builtin_prefetch(mrowp + (size_t)(half * 8) * R_RES + r0 + 32, 0, 1);

    // ---- logits: two 16x16 tiles, padded K = 64 -> 2 WMMA steps each
    v8f c0 = {}, c1 = {};
    c0 = WMMA_F16(aq0, load_b_tr(kh, C_PAD, 0,  r0),      c0);
    c0 = WMMA_F16(aq1, load_b_tr(kh, C_PAD, 32, r0),      c0);
    c1 = WMMA_F16(aq0, load_b_tr(kh, C_PAD, 0,  r0 + 16), c1);
    c1 = WMMA_F16(aq1, load_b_tr(kh, C_PAD, 32, r0 + 16), c1);

    // ---- mask bias + online softmax (rows live across 16-lane groups)
#pragma unroll
    for (int i = 0; i < 8; ++i) {
      const int row = half * 8 + i;
      c0[i] += BIGF * (mrowp[(size_t)row * R_RES + r0 + n]      - 1.f);
      c1[i] += BIGF * (mrowp[(size_t)row * R_RES + r0 + 16 + n] - 1.f);

      float mt = fmaxf(c0[i], c1[i]);
#pragma unroll
      for (int off = 8; off; off >>= 1) mt = fmaxf(mt, __shfl_xor(mt, off, 16));
      const float mnew = fmaxf(mrow[i], mt);
      const float scl  = __expf(mrow[i] - mnew);
      const float p0 = __expf(c0[i] - mnew);
      const float p1 = __expf(c1[i] - mnew);
      float rs = p0 + p1;
#pragma unroll
      for (int off = 8; off; off >>= 1) rs += __shfl_xor(rs, off, 16);
      lrow[i] = lrow[i] * scl + rs;
      mrow[i] = mnew;
      o0[i] *= scl; o1[i] *= scl; o2[i] *= scl;
      ptile[row * 32 + n]      = (_Float16)p0;
      ptile[row * 32 + 16 + n] = (_Float16)p1;
    }
    __syncthreads();                             // order LDS store -> A-frag load

    // ---- P @ V chunk: A = P (16x32), B = V[r0:r0+32, h*48 + {0,16,32}]
    const v16h ap = load_a_rm(ptile, 32, 0, 0);
    o0 = WMMA_F16(ap, load_b_tr(vTh, R_RES, r0, 0),  o0);
    o1 = WMMA_F16(ap, load_b_tr(vTh, R_RES, r0, 16), o1);
    o2 = WMMA_F16(ap, load_b_tr(vTh, R_RES, r0, 32), o2);
    __syncthreads();                             // guard ptile reuse next iter
  }

  // ---- normalize and write weighted average
#pragma unroll
  for (int i = 0; i < 8; ++i) {
    const float inv = 1.f / lrow[i];
    const size_t row = (size_t)(t0 + half * 8 + i) * C_DIM + h * KD + n;
    outw[row]      = o0[i] * inv;
    outw[row + 16] = o1[i] * inv;
    outw[row + 32] = o2[i] * inv;
  }
}

// ---------------------------------------------------------------------------
// Elementwise helpers
// ---------------------------------------------------------------------------
__global__ void f32_to_f16_tr_kernel(const float* __restrict__ in,  // [K, N]
                                     _Float16* __restrict__ outT,   // [N, K]
                                     int K, int N) {
  const int i = blockIdx.x * blockDim.x + threadIdx.x;
  if (i < K * N) {
    const int k = i / N, n = i % N;
    outT[(size_t)n * K + k] = (_Float16)in[i];
  }
}

// head-pad repack: [rows, 384] -> [rows, 512], cols h*64+c = src h*48+c, else 0
__global__ void pad_head_kernel(const _Float16* __restrict__ src,
                                _Float16* __restrict__ dst, int rows) {
  const int i = blockIdx.x * blockDim.x + threadIdx.x;
  if (i < rows * C_PAD) {
    const int r = i / C_PAD, col = i % C_PAD;
    const int h = col >> 6, c = col & 63;
    dst[i] = (c < KD) ? src[(size_t)r * C_DIM + h * KD + c] : (_Float16)0.f;
  }
}

__global__ void transpose_h_kernel(const _Float16* __restrict__ src,
                                   _Float16* __restrict__ dst, int R, int C) {
  const int i = blockIdx.x * blockDim.x + threadIdx.x;
  if (i < R * C) {
    const int r = i / C, c = i % C;
    dst[(size_t)c * R + r] = src[i];
  }
}

__global__ void gate_mul_kernel(const float* __restrict__ wavg,
                                const float* __restrict__ gacc,
                                _Float16* __restrict__ out, int nelem) {
  const int i = blockIdx.x * blockDim.x + threadIdx.x;
  if (i < nelem) {
    const float g = 1.f / (1.f + __expf(-gacc[i]));
    out[i] = (_Float16)(wavg[i] * g);
  }
}

// ---------------------------------------------------------------------------
// Host orchestration
// ---------------------------------------------------------------------------
extern "C" void kernel_launch(void* const* d_in, const int* in_sizes, int n_in,
                              void* d_out, int out_size, void* d_ws, size_t ws_size,
                              hipStream_t stream) {
  (void)in_sizes; (void)n_in; (void)out_size; (void)ws_size;
  const float* original  = (const float*)d_in[0];
  const float* resampled = (const float*)d_in[1];
  const float* attn_mask = (const float*)d_in[2];
  const float* qn_scale  = (const float*)d_in[5];
  const float* qn_offset = (const float*)d_in[6];
  const float* dn_scale  = (const float*)d_in[7];
  const float* dn_offset = (const float*)d_in[8];
  const float* query_w   = (const float*)d_in[9];
  const float* key_w     = (const float*)d_in[10];
  const float* value_w   = (const float*)d_in[11];
  const float* gating_w  = (const float*)d_in[12];
  const float* gating_b  = (const float*)d_in[13];
  const float* output_w  = (const float*)d_in[14];
  const float* output_b  = (const float*)d_in[15];
  const float* rt_ln_s   = (const float*)d_in[16];
  const float* rt_ln_o   = (const float*)d_in[17];
  const float* rt_w1     = (const float*)d_in[18];
  const float* rt_b1     = (const float*)d_in[19];
  const float* rt_w2     = (const float*)d_in[20];
  const float* rt_b2     = (const float*)d_in[21];
  const float* ot_ln_s   = (const float*)d_in[22];
  const float* ot_ln_o   = (const float*)d_in[23];
  const float* ot_w1     = (const float*)d_in[24];
  const float* ot_b1     = (const float*)d_in[25];
  const float* ot_w2     = (const float*)d_in[26];
  const float* ot_b2     = (const float*)d_in[27];

  float* out_res = (float*)d_out;                          // [512, 384]
  float* out_org = (float*)d_out + (size_t)T_TOK * C_DIM;  // [4096, 384]

  // bump allocator over workspace (256B aligned chunks)
  char* wsp = (char*)d_ws;
  auto alloc_h = [&](size_t ne) { _Float16* p = (_Float16*)wsp;
                                  wsp += ((ne * 2 + 255) / 256) * 256; return p; };
  auto alloc_f = [&](size_t ne) { float* p = (float*)wsp;
                                  wsp += ((ne * 4 + 255) / 256) * 256; return p; };

  _Float16* qin16  = alloc_h((size_t)T_TOK * C_DIM);
  _Float16* din16  = alloc_h((size_t)R_RES * C_DIM);
  _Float16* WqT16  = alloc_h((size_t)C_DIM * C_DIM);   // all weights stored [N,K]
  _Float16* WkT16  = alloc_h((size_t)C_DIM * C_DIM);
  _Float16* WvT16  = alloc_h((size_t)C_DIM * C_DIM);
  _Float16* WgT16  = alloc_h((size_t)C_DIM * C_DIM);
  _Float16* WoT16  = alloc_h((size_t)C_DIM * C_DIM);
  _Float16* W1rT16 = alloc_h((size_t)F_DIM * C_DIM);
  _Float16* W2rT16 = alloc_h((size_t)C_DIM * F_DIM);
  _Float16* W1oT16 = alloc_h((size_t)F_DIM * C_DIM);
  _Float16* W2oT16 = alloc_h((size_t)C_DIM * F_DIM);
  _Float16* q16    = alloc_h((size_t)T_TOK * C_DIM);
  _Float16* k16    = alloc_h((size_t)R_RES * C_DIM);
  _Float16* v16    = alloc_h((size_t)R_RES * C_DIM);
  _Float16* qpad16 = alloc_h((size_t)T_TOK * C_PAD);
  _Float16* kpad16 = alloc_h((size_t)R_RES * C_PAD);
  _Float16* vT16   = alloc_h((size_t)C_DIM * R_RES);
  _Float16* gated16 = alloc_h((size_t)T_TOK * C_DIM);
  _Float16* rln16  = alloc_h((size_t)T_TOK * C_DIM);
  _Float16* hres16 = alloc_h((size_t)T_TOK * F_DIM);
  _Float16* oln16  = alloc_h((size_t)R_RES * C_DIM);
  _Float16* ho16   = alloc_h((size_t)R_RES * F_DIM);
  float* gateacc   = alloc_f((size_t)T_TOK * C_DIM);
  float* weighted  = alloc_f((size_t)T_TOK * C_DIM);
  float* res1      = alloc_f((size_t)T_TOK * C_DIM);

  auto cvtT = [&](const float* src, _Float16* dstT, int K, int N) {
    const int ne = K * N;
    f32_to_f16_tr_kernel<<<(ne + 255) / 256, 256, 0, stream>>>(src, dstT, K, N);
  };
  auto gemm = [&](const _Float16* A, int lda, const _Float16* BT,
                  const float* bias, const float* resid, float alpha, int relu,
                  float* oF, _Float16* oH, int M, int N, int K) {
    const int waves = (M / 16) * (N / 64);
    gemm_f16_kernel<<<(waves + 7) / 8, 256, 0, stream>>>(
        A, lda, BT, bias, resid, alpha, relu, oF, oH, M, N, K);
  };

  // 1) weights -> f16, transposed to [N, K]
  cvtT(query_w, WqT16, C_DIM, C_DIM);
  cvtT(key_w,   WkT16, C_DIM, C_DIM);
  cvtT(value_w, WvT16, C_DIM, C_DIM);
  cvtT(gating_w,WgT16, C_DIM, C_DIM);
  cvtT(output_w,WoT16, C_DIM, C_DIM);
  cvtT(rt_w1,   W1rT16, C_DIM, F_DIM);
  cvtT(rt_w2,   W2rT16, F_DIM, C_DIM);
  cvtT(ot_w1,   W1oT16, C_DIM, F_DIM);
  cvtT(ot_w2,   W2oT16, F_DIM, C_DIM);

  // 2) LayerNorms for attention inputs
  layernorm_f16_kernel<<<T_TOK, 128, 0, stream>>>(resampled, qn_scale, qn_offset, qin16, C_DIM);
  layernorm_f16_kernel<<<R_RES, 128, 0, stream>>>(original,  dn_scale, dn_offset, din16, C_DIM);

  // 3) projections (q pre-scaled by kd^-0.5)
  const float qscale = 0.14433756729740643f;  // 48^-0.5
  gemm(qin16, C_DIM, WqT16, nullptr, nullptr, qscale, 0, nullptr, q16, T_TOK, C_DIM, C_DIM);
  gemm(din16, C_DIM, WkT16, nullptr, nullptr, 1.f,    0, nullptr, k16, R_RES, C_DIM, C_DIM);
  gemm(din16, C_DIM, WvT16, nullptr, nullptr, 1.f,    0, nullptr, v16, R_RES, C_DIM, C_DIM);
  gemm(qin16, C_DIM, WgT16, gating_b, nullptr, 1.f,   0, gateacc, nullptr, T_TOK, C_DIM, C_DIM);

  // 4) head-pad Q/K (kd 48 -> 64) and transpose V for contiguous B-fragments
  pad_head_kernel<<<((T_TOK * C_PAD) + 255) / 256, 256, 0, stream>>>(q16, qpad16, T_TOK);
  pad_head_kernel<<<((R_RES * C_PAD) + 255) / 256, 256, 0, stream>>>(k16, kpad16, R_RES);
  transpose_h_kernel<<<((R_RES * C_DIM) + 255) / 256, 256, 0, stream>>>(v16, vT16, R_RES, C_DIM);

  // 5) flash attention
  flash_attn_kernel<<<dim3(T_TOK / 16, H_HEAD), 32, 0, stream>>>(
      qpad16, kpad16, vT16, attn_mask, weighted);

  // 6) sigmoid gate * weighted -> f16
  gate_mul_kernel<<<((T_TOK * C_DIM) + 255) / 256, 256, 0, stream>>>(
      weighted, gateacc, gated16, T_TOK * C_DIM);

  // 7) output projection + bias + residual -> res1 (resampled after attention)
  gemm(gated16, C_DIM, WoT16, output_b, resampled, 1.f, 0, res1, nullptr, T_TOK, C_DIM, C_DIM);

  // 8) resampled transition
  layernorm_f16_kernel<<<T_TOK, 128, 0, stream>>>(res1, rt_ln_s, rt_ln_o, rln16, C_DIM);
  gemm(rln16, C_DIM, W1rT16, rt_b1, nullptr, 1.f, 1, nullptr, hres16, T_TOK, F_DIM, C_DIM);
  gemm(hres16, F_DIM, W2rT16, rt_b2, res1,   1.f, 0, out_res, nullptr, T_TOK, C_DIM, F_DIM);

  // 9) original transition
  layernorm_f16_kernel<<<R_RES, 128, 0, stream>>>(original, ot_ln_s, ot_ln_o, oln16, C_DIM);
  gemm(oln16, C_DIM, W1oT16, ot_b1, nullptr,  1.f, 1, nullptr, ho16, R_RES, F_DIM, C_DIM);
  gemm(ho16,  F_DIM, W2oT16, ot_b2, original, 1.f, 0, out_org, nullptr, R_RES, C_DIM, F_DIM);
}
